// HierarchicalTransformerBlock_62612033241275
// MI455X (gfx1250) — matrix operand order repaired and checked by reading, not compiled, not run on hardware
//
#include <hip/hip_runtime.h>
#include <hip/hip_bf16.h>
#include <math.h>

// ---------------- problem constants ----------------
#define Bc   2
#define Nc   512
#define Tc   128
#define Dc   256
#define Hc   8
#define Hdc  32
#define FFc  1024
#define RFc  5
#define RBHc 64
#define RVHc 128
#define BNc  (Bc*Nc)        // 1024
#define BNTc (BNc*Tc)       // 131072

typedef __attribute__((ext_vector_type(16))) __bf16 v16bf;
typedef __attribute__((ext_vector_type(8)))  float  v8f;
typedef __attribute__((ext_vector_type(4)))  unsigned int v4u;
struct Frag32 { v4u lo; v4u hi; };   // 32 bytes == v16bf

// ---------------- block reduction helpers (blockDim.x == 256) ----------------
__device__ __forceinline__ float block_reduce_sum(float v, float* red) {
    int tid = threadIdx.x;
    red[tid] = v; __syncthreads();
    #pragma unroll
    for (int s = 128; s > 0; s >>= 1) {
        if (tid < s) red[tid] += red[tid + s];
        __syncthreads();
    }
    float r = red[0]; __syncthreads();
    return r;
}
__device__ __forceinline__ float block_reduce_max(float v, float* red) {
    int tid = threadIdx.x;
    red[tid] = v; __syncthreads();
    #pragma unroll
    for (int s = 128; s > 0; s >>= 1) {
        if (tid < s) red[tid] = fmaxf(red[tid], red[tid + s]);
        __syncthreads();
    }
    float r = red[0]; __syncthreads();
    return r;
}
__device__ __forceinline__ float gelu_exact(float x) {
    return 0.5f * x * (1.0f + erff(x * 0.70710678118654752f));
}

// ---------------- K1: masked summary + pos_sum ----------------
__global__ __launch_bounds__(256) void k_summary(
    const float* __restrict__ pb, const int* __restrict__ masks,
    const float* __restrict__ pos, float* __restrict__ s, float* __restrict__ pos_sum)
{
    int bn = blockIdx.x, tid = threadIdx.x;
    __shared__ float msh[Tc];
    __shared__ float inv_sh;
    if (tid < Tc) msh[tid] = (float)masks[(size_t)bn * Tc + tid];
    __syncthreads();
    if (tid == 0) {
        float c = 0.f;
        for (int t = 0; t < Tc; t++) c += msh[t];
        inv_sh = 1.0f / fmaxf(c, 1e-6f);
    }
    __syncthreads();
    float inv = inv_sh;
    const float* p = pb + (size_t)bn * Tc * Dc + tid;
    float acc = 0.f;
    for (int t = 0; t < Tc; t++) acc += p[(size_t)t * Dc] * msh[t];
    s[(size_t)bn * Dc + tid] = acc * inv;
    if (tid < 3) {
        const float* pp = pos + (size_t)bn * Tc * 3 + tid;
        float a = 0.f;
        for (int t = 0; t < Tc; t++) a += pp[t * 3] * msh[t];
        pos_sum[bn * 3 + tid] = a * inv;
    }
}

// ---------------- f32 -> bf16 conversion ----------------
__global__ __launch_bounds__(256) void k_cvt(const float* __restrict__ in,
                                             __bf16* __restrict__ out, int n)
{
    int i = blockIdx.x * 256 + threadIdx.x;
    if (i < n) out[i] = (__bf16)in[i];
}

// ---------------- generic WMMA bf16 GEMM: C = act(A@B + bias) ----------------
// A [M,K] bf16 row-major, B [K,N] bf16 row-major; M%64==0, N%64==0, K%32==0.
// Block: 256 threads = 8 waves; block tile 64x64. Each wave owns a 16x32 strip:
// one A fragment feeds two v_wmma_f32_16x16x32_bf16 (B cols tn2, tn2+16).
// A tile staged global->LDS with GLOBAL_LOAD_ASYNC_TO_LDS_B128 (ASYNCcnt).
// B tile stored transposed (BsT[n][k]) so all fragments are contiguous 16B reads.
#define LDA  40
#define LDBT 40
template <int ACT, int OBF>
__global__ __launch_bounds__(256) void k_gemm(
    const __bf16* __restrict__ A, const __bf16* __restrict__ Bm,
    const float* __restrict__ bias, float* __restrict__ Of, __bf16* __restrict__ Ob,
    int M, int Nn, int K)
{
    __shared__ alignas(16) __bf16 As[64 * LDA];    // 64 rows x 32 k
    __shared__ alignas(16) __bf16 BsT[64 * LDBT];  // 64 cols x 32 k (transposed)
    int tid = threadIdx.x, lane = tid & 31, w = tid >> 5;
    int row0 = blockIdx.y * 64, col0 = blockIdx.x * 64;
    int tm = (w >> 1) * 16, tn2 = (w & 1) * 32;
    // A-tile copy assignment: thread -> (row ar, 8 cols at ac), 16 bytes per lane
    int ar = tid >> 2, ac = (tid & 3) * 8;
    unsigned lds_a = (unsigned)(size_t)(void*)(As + ar * LDA + ac);   // LDS byte offset
    const __bf16* asrc = A + (size_t)(row0 + ar) * K + ac;
    // B-tile copy assignment: thread -> (k row br, 8 cols at bc)
    int br = tid >> 3, bc = (tid & 7) * 8;
    const __bf16* bsrc = Bm + (size_t)br * Nn + col0 + bc;
    const size_t bstep = (size_t)32 * Nn;
    v8f acc0 = {}, acc1 = {};
    for (int k0 = 0; k0 < K; k0 += 32) {
        // ---- A tile 64x32: async DMA global -> LDS, one B128 per lane ----
        asm volatile("global_load_async_to_lds_b128 %0, %1, off"
                     :: "v"(lds_a), "v"(asrc) : "memory");
        asrc += 32;
        // ---- B tile 32x64, transposed into BsT ----
        {
            if (k0 + 32 < K) __builtin_prefetch(bsrc + bstep, 0, 1);
            __bf16 tmp[8];
            #pragma unroll
            for (int i = 0; i < 8; i++) tmp[i] = bsrc[i];
            #pragma unroll
            for (int i = 0; i < 8; i++) BsT[(bc + i) * LDBT + br] = tmp[i];
            bsrc += bstep;
        }
        asm volatile("s_wait_asynccnt 0x0" ::: "memory");
        __syncthreads();
        // ---- fragments (contiguous 16B LDS reads) ----
        // A: lane m = lane&15; K-half kh = (lane>>4)*8; elems 0..7 -> kh+i, 8..15 -> 16+kh+i
        int m = tm + (lane & 15);
        int kh = (lane >> 4) * 8;
        Frag32 fa;
        fa.lo = *(const v4u*)(As + m * LDA + kh);
        fa.hi = *(const v4u*)(As + m * LDA + 16 + kh);
        v16bf a = __builtin_bit_cast(v16bf, fa);
        // B: col n = lane&15; K-half khb = (lane>>4)*8; elems 0..7 -> khb+i, 8..15 -> 16+khb+i
        int n = tn2 + (lane & 15);
        int khb = (lane >> 4) * 8;
        Frag32 fb0, fb1;
        fb0.lo = *(const v4u*)(BsT + n * LDBT + khb);
        fb0.hi = *(const v4u*)(BsT + n * LDBT + khb + 16);
        fb1.lo = *(const v4u*)(BsT + (n + 16) * LDBT + khb);
        fb1.hi = *(const v4u*)(BsT + (n + 16) * LDBT + khb + 16);
        v16bf b0 = __builtin_bit_cast(v16bf, fb0);
        v16bf b1 = __builtin_bit_cast(v16bf, fb1);
        acc0 = __builtin_amdgcn_wmma_f32_16x16x32_bf16(false, a, false, b0,
                                                       (short)0, acc0, false, false);
        acc1 = __builtin_amdgcn_wmma_f32_16x16x32_bf16(false, a, false, b1,
                                                       (short)0, acc1, false, false);
        __syncthreads();
    }
    // epilogue: C/D layout m = r + 8*(lane>>4), n = lane&15
    int n = lane & 15;
    int colA = col0 + tn2 + n;
    int colB = colA + 16;
    float bvA = bias ? bias[colA] : 0.f;
    float bvB = bias ? bias[colB] : 0.f;
    #pragma unroll
    for (int r = 0; r < 8; r++) {
        int m = tm + r + 8 * (lane >> 4);
        size_t rowb = (size_t)(row0 + m) * Nn;
        float vA = acc0[r] + bvA;
        float vB = acc1[r] + bvB;
        if (ACT == 1) { vA = fmaxf(vA, 0.f); vB = fmaxf(vB, 0.f); }
        else if (ACT == 2) { vA = 1.f / (1.f + expf(-vA)); vB = 1.f / (1.f + expf(-vB)); }
        if constexpr (OBF) { Ob[rowb + colA] = (__bf16)vA; Ob[rowb + colB] = (__bf16)vB; }
        else               { Of[rowb + colA] = vA;        Of[rowb + colB] = vB; }
    }
}

// ---------------- K4: attention logits ----------------
__global__ __launch_bounds__(256) void k_logits(
    const float* __restrict__ q, const float* __restrict__ k, float* __restrict__ logits)
{
    int idx = blockIdx.x;                 // b*H*N + h*N + qi
    int qi = idx & (Nc - 1), h = (idx >> 9) & 7, b = idx >> 12;
    __shared__ float qs[Hdc];
    int tid = threadIdx.x;
    if (tid < Hdc) qs[tid] = q[(size_t)(b * Nc + qi) * Dc + h * Hdc + tid];
    __syncthreads();
    const float scale = 0.17677669529663689f;   // 1/sqrt(32)
    for (int kk = tid; kk < Nc; kk += 256) {
        const float* kp = k + (size_t)(b * Nc + kk) * Dc + h * Hdc;
        float acc = 0.f;
        #pragma unroll
        for (int j = 0; j < Hdc; j++) acc += qs[j] * kp[j];
        logits[((size_t)(b * Hc + h) * Nc + qi) * Nc + kk] = acc * scale;
    }
}

// ---------------- K5: fused relative-bias MLP (+= into logits) ----------------
__global__ __launch_bounds__(256) void k_relbias(
    const float* __restrict__ pos_sum, const float* __restrict__ Wb1,
    const float* __restrict__ bb1, const float* __restrict__ Wb2,
    const float* __restrict__ bb2, float* __restrict__ logits)
{
    __shared__ float w1[RFc * RBHc];
    __shared__ float w2[RBHc * Hc];
    __shared__ float b1[RBHc];
    int tid = threadIdx.x;
    for (int i = tid; i < RFc * RBHc; i += 256) w1[i] = Wb1[i];
    for (int i = tid; i < RBHc * Hc; i += 256) w2[i] = Wb2[i];
    if (tid < RBHc) b1[tid] = bb1[tid];
    __syncthreads();
    size_t gid = (size_t)blockIdx.x * 256 + tid;     // b*N*N + q*N + k
    int kk = gid & (Nc - 1), qi = (gid >> 9) & (Nc - 1), b = (int)(gid >> 18);
    const float* pq = pos_sum + (b * Nc + qi) * 3;
    const float* pk = pos_sum + (b * Nc + kk) * 3;
    float d0 = pq[0] - pk[0], d1 = pq[1] - pk[1], d2 = pq[2] - pk[2];
    float dist2 = d0 * d0 + d1 * d1 + d2 * d2;
    float dist = sqrtf(dist2);
    float f[RFc] = {d0, d1, d2, dist, dist2};
    float outh[Hc] = {};
    for (int j = 0; j < RBHc; j++) {
        float hv = b1[j];
        #pragma unroll
        for (int r = 0; r < RFc; r++) hv += f[r] * w1[r * RBHc + j];
        hv = gelu_exact(hv);
        #pragma unroll
        for (int h = 0; h < Hc; h++) outh[h] += hv * w2[j * Hc + h];
    }
    #pragma unroll
    for (int h = 0; h < Hc; h++) {
        size_t li = ((size_t)(b * Hc + h) * Nc + qi) * Nc + kk;
        logits[li] += outh[h] + bb2[h];
    }
}

// ---------------- K6: softmax over last axis (N=512), in place ----------------
__global__ __launch_bounds__(256) void k_softmax(float* __restrict__ lg)
{
    __shared__ float red[256];
    int tid = threadIdx.x;
    size_t base = (size_t)blockIdx.x * Nc;
    float a = lg[base + tid], b = lg[base + tid + 256];
    float m = block_reduce_max(fmaxf(a, b), red);
    float e1 = expf(a - m), e2 = expf(b - m);
    float ssum = block_reduce_sum(e1 + e2, red);
    float inv = 1.f / ssum;
    lg[base + tid] = e1 * inv;
    lg[base + tid + 256] = e2 * inv;
}

// ---------------- K7: attn = aw @ v ----------------
__global__ __launch_bounds__(256) void k_attn1(
    const float* __restrict__ aw, const float* __restrict__ v, float* __restrict__ attn)
{
    __shared__ float awsh[Hc * Nc];   // 16 KB
    int bn = blockIdx.x, b = bn >> 9, qi = bn & (Nc - 1), tid = threadIdx.x;
    for (int i = tid; i < Hc * Nc; i += 256)
        awsh[i] = aw[((size_t)(b * Hc + (i >> 9)) * Nc + qi) * Nc + (i & (Nc - 1))];
    __syncthreads();
    int h = tid >> 5;
    float acc = 0.f;
    for (int kk = 0; kk < Nc; kk++)
        acc += awsh[h * Nc + kk] * v[(size_t)(b * Nc + kk) * Dc + tid];
    attn[(size_t)bn * Dc + tid] = acc;
}

// ---------------- K8: fused rel-value einsum (wsum trick), += into attn ----------------
__global__ __launch_bounds__(128) void k_attnrel(
    const float* __restrict__ aw, const float* __restrict__ pos_sum,
    const float* __restrict__ Wv1, const float* __restrict__ bv1,
    const float* __restrict__ Wv2, const float* __restrict__ bv2,
    float* __restrict__ attn)
{
    __shared__ float awsh[Hc * Nc];    // 16 KB
    __shared__ float posk[Nc * 3];     // 6 KB
    __shared__ float w1s[RFc * RVHc];  // 2.5 KB
    __shared__ float b1s[RVHc];
    __shared__ float wsum[Hc * RVHc];  // 4 KB
    int bn = blockIdx.x, b = bn >> 9, qi = bn & (Nc - 1), tid = threadIdx.x;
    for (int i = tid; i < Hc * Nc; i += 128)
        awsh[i] = aw[((size_t)(b * Hc + (i >> 9)) * Nc + qi) * Nc + (i & (Nc - 1))];
    for (int i = tid; i < Nc * 3; i += 128) posk[i] = pos_sum[(size_t)b * Nc * 3 + i];
    for (int i = tid; i < RFc * RVHc; i += 128) w1s[i] = Wv1[i];
    if (tid < RVHc) b1s[tid] = bv1[tid];
    __syncthreads();
    float p0 = posk[qi * 3 + 0], p1 = posk[qi * 3 + 1], p2 = posk[qi * 3 + 2];
    int r = tid;
    float acc[Hc] = {};
    for (int kk = 0; kk < Nc; kk++) {
        float d0 = p0 - posk[kk * 3 + 0];
        float d1 = p1 - posk[kk * 3 + 1];
        float d2 = p2 - posk[kk * 3 + 2];
        float dist2 = d0 * d0 + d1 * d1 + d2 * d2;
        float dist = sqrtf(dist2);
        float hv = b1s[r] + d0 * w1s[0 * RVHc + r] + d1 * w1s[1 * RVHc + r]
                 + d2 * w1s[2 * RVHc + r] + dist * w1s[3 * RVHc + r]
                 + dist2 * w1s[4 * RVHc + r];
        hv = gelu_exact(hv);
        #pragma unroll
        for (int h = 0; h < Hc; h++) acc[h] += awsh[h * Nc + kk] * hv;
    }
    #pragma unroll
    for (int h = 0; h < Hc; h++) wsum[h * RVHc + r] = acc[h];
    __syncthreads();
    // stage 2: attn_rel[o] = bv2[o] + sum_r wsum[h][r]*Wv2[r, o]   (softmax rows sum to 1)
    for (int o = tid; o < Dc; o += 128) {
        int h = o >> 5;
        float a2 = bv2[o];
        for (int rr = 0; rr < RVHc; rr++) a2 += wsum[h * RVHc + rr] * Wv2[(size_t)rr * Dc + o];
        attn[(size_t)bn * Dc + o] += a2;
    }
}

// ---------------- K9: updated_summary + glob_n; write output #2 ----------------
__global__ __launch_bounds__(256) void k_us(
    const float* __restrict__ attn, const float* __restrict__ s,
    float* __restrict__ us, float* __restrict__ gn, float* __restrict__ dout2)
{
    __shared__ float red[256];
    int bn = blockIdx.x, tid = threadIdx.x;
    float u = attn[(size_t)bn * Dc + tid] + s[(size_t)bn * Dc + tid];
    us[(size_t)bn * Dc + tid] = u;
    dout2[(size_t)bn * Dc + tid] = u;
    float n2 = block_reduce_sum(u * u, red);
    gn[(size_t)bn * Dc + tid] = u / fmaxf(sqrtf(n2), 1e-6f);
}

// ---------------- K10: combined = [local_n | glob_n] in bf16 ----------------
__global__ __launch_bounds__(256) void k_combined(
    const float* __restrict__ pb, const float* __restrict__ gn, __bf16* __restrict__ comb)
{
    __shared__ float red[256];
    size_t row = blockIdx.x;            // bn*T + t
    int bn = (int)(row >> 7), tid = threadIdx.x;
    float xv = pb[row * Dc + tid];
    float n2 = block_reduce_sum(xv * xv, red);
    float inv = 1.f / fmaxf(sqrtf(n2), 1e-6f);
    __bf16* cr = comb + row * (2 * Dc);
    cr[tid]      = (__bf16)(xv * inv);
    cr[Dc + tid] = (__bf16)gn[(size_t)bn * Dc + tid];
}

// ---------------- K12: LayerNorm(gate) then x = (1-g)*pb + g*us ----------------
__global__ __launch_bounds__(256) void k_lnx(
    float* __restrict__ gate_x, const float* __restrict__ pb, const float* __restrict__ us,
    const float* __restrict__ gn_g, const float* __restrict__ gn_b, __bf16* __restrict__ xbf)
{
    __shared__ float red[256];
    size_t row = blockIdx.x;
    int bn = (int)(row >> 7), tid = threadIdx.x;
    float g = gate_x[row * Dc + tid];
    float mean = block_reduce_sum(g, red) * (1.f / Dc);
    float d = g - mean;
    float var = block_reduce_sum(d * d, red) * (1.f / Dc);
    float gate = d * rsqrtf(var + 1e-5f) * gn_g[tid] + gn_b[tid];
    float xv = (1.f - gate) * pb[row * Dc + tid] + gate * us[(size_t)bn * Dc + tid];
    gate_x[row * Dc + tid] = xv;        // overwrite in place: buffer now holds x
    xbf[row * Dc + tid] = (__bf16)xv;
}

// ---------------- K15: out = LayerNorm(h + x) in place on d_out ----------------
__global__ __launch_bounds__(256) void k_lnfinal(
    float* __restrict__ dout, const float* __restrict__ x,
    const float* __restrict__ fn_g, const float* __restrict__ fn_b)
{
    __shared__ float red[256];
    size_t row = blockIdx.x;
    int tid = threadIdx.x;
    float hv = dout[row * Dc + tid] + x[row * Dc + tid];
    float mean = block_reduce_sum(hv, red) * (1.f / Dc);
    float d = hv - mean;
    float var = block_reduce_sum(d * d, red) * (1.f / Dc);
    dout[row * Dc + tid] = d * rsqrtf(var + 1e-5f) * fn_g[tid] + fn_b[tid];
}

// =====================================================================
extern "C" void kernel_launch(void* const* d_in, const int* in_sizes, int n_in,
                              void* d_out, int out_size, void* d_ws, size_t ws_size,
                              hipStream_t stream)
{
    const float* pb   = (const float*)d_in[0];
    const int*   mk   = (const int*)d_in[2];
    const float* pos  = (const float*)d_in[3];
    const float* Wq = (const float*)d_in[4],  *bq = (const float*)d_in[5];
    const float* Wk = (const float*)d_in[6],  *bk = (const float*)d_in[7];
    const float* Wv = (const float*)d_in[8],  *bv = (const float*)d_in[9];
    const float* Wb1 = (const float*)d_in[10], *bb1 = (const float*)d_in[11];
    const float* Wb2 = (const float*)d_in[12], *bb2 = (const float*)d_in[13];
    const float* Wv1 = (const float*)d_in[14], *bv1 = (const float*)d_in[15];
    const float* Wv2 = (const float*)d_in[16], *bv2 = (const float*)d_in[17];
    const float* Wg = (const float*)d_in[18],  *bg = (const float*)d_in[19];
    const float* gng = (const float*)d_in[20], *gnb = (const float*)d_in[21];
    const float* Wf1 = (const float*)d_in[22], *bf1 = (const float*)d_in[23];
    const float* Wf2 = (const float*)d_in[24], *bf2 = (const float*)d_in[25];
    const float* fng = (const float*)d_in[26], *fnb = (const float*)d_in[27];

    char* wsb = (char*)d_ws;
    size_t off = 0;
    auto alloc = [&](size_t bytes) { size_t o = off; off = (off + bytes + 255) & ~(size_t)255; return o; };

    float*  s_f    = (float*)(wsb + alloc((size_t)BNc * Dc * 4));
    __bf16* s_bf   = (__bf16*)(wsb + alloc((size_t)BNc * Dc * 2));
    float*  pos_f  = (float*)(wsb + alloc((size_t)BNc * 3 * 4));
    float*  q_f    = (float*)(wsb + alloc((size_t)BNc * Dc * 4));
    float*  kk_f   = (float*)(wsb + alloc((size_t)BNc * Dc * 4));
    float*  v_f    = (float*)(wsb + alloc((size_t)BNc * Dc * 4));
    float*  attn_f = (float*)(wsb + alloc((size_t)BNc * Dc * 4));
    float*  us_f   = (float*)(wsb + alloc((size_t)BNc * Dc * 4));
    float*  gn_f   = (float*)(wsb + alloc((size_t)BNc * Dc * 4));
    __bf16* wq_bf  = (__bf16*)(wsb + alloc((size_t)Dc * Dc * 2));
    __bf16* wk_bf  = (__bf16*)(wsb + alloc((size_t)Dc * Dc * 2));
    __bf16* wv_bf  = (__bf16*)(wsb + alloc((size_t)Dc * Dc * 2));
    __bf16* wg_bf  = (__bf16*)(wsb + alloc((size_t)2 * Dc * Dc * 2));
    __bf16* wf1_bf = (__bf16*)(wsb + alloc((size_t)Dc * FFc * 2));
    __bf16* wf2_bf = (__bf16*)(wsb + alloc((size_t)FFc * Dc * 2));
    float*  logits = (float*)(wsb + alloc((size_t)Bc * Hc * Nc * Nc * 4));
    float*  gate_x = (float*)(wsb + alloc((size_t)BNTc * Dc * 4));
    __bf16* x_bf   = (__bf16*)(wsb + alloc((size_t)BNTc * Dc * 2));
    // lifetime union: combined (gate GEMM input) dies before h1 (FF1 output) is written
    char*   unionp = wsb + alloc((size_t)BNTc * FFc * 2);
    __bf16* comb_bf = (__bf16*)unionp;   // BNT x 2D bf16 (first half of union)
    __bf16* h1_bf   = (__bf16*)unionp;   // BNT x FF bf16

    float* dout1 = (float*)d_out;                       // (B,N,T,D)
    float* dout2 = (float*)d_out + (size_t)BNTc * Dc;   // (B,N,1,D)

    // 1. summary + pos_sum
    k_summary<<<BNc, 256, 0, stream>>>(pb, mk, pos, s_f, pos_f);
    // 2. bf16 conversions
    k_cvt<<<(BNc * Dc + 255) / 256, 256, 0, stream>>>(s_f, s_bf, BNc * Dc);
    k_cvt<<<(Dc * Dc + 255) / 256, 256, 0, stream>>>(Wq, wq_bf, Dc * Dc);
    k_cvt<<<(Dc * Dc + 255) / 256, 256, 0, stream>>>(Wk, wk_bf, Dc * Dc);
    k_cvt<<<(Dc * Dc + 255) / 256, 256, 0, stream>>>(Wv, wv_bf, Dc * Dc);
    k_cvt<<<(2 * Dc * Dc + 255) / 256, 256, 0, stream>>>(Wg, wg_bf, 2 * Dc * Dc);
    k_cvt<<<(Dc * FFc + 255) / 256, 256, 0, stream>>>(Wf1, wf1_bf, Dc * FFc);
    k_cvt<<<(FFc * Dc + 255) / 256, 256, 0, stream>>>(Wf2, wf2_bf, FFc * Dc);
    // 3. QKV projections (WMMA)
    dim3 gqkv(Dc / 64, BNc / 64);
    k_gemm<0, 0><<<gqkv, 256, 0, stream>>>(s_bf, wq_bf, bq, q_f, nullptr, BNc, Dc, Dc);
    k_gemm<0, 0><<<gqkv, 256, 0, stream>>>(s_bf, wk_bf, bk, kk_f, nullptr, BNc, Dc, Dc);
    k_gemm<0, 0><<<gqkv, 256, 0, stream>>>(s_bf, wv_bf, bv, v_f, nullptr, BNc, Dc, Dc);
    // 4-6. logits + relative bias + softmax
    k_logits<<<Bc * Hc * Nc, 256, 0, stream>>>(q_f, kk_f, logits);
    k_relbias<<<(Bc * Nc * Nc) / 256, 256, 0, stream>>>(pos_f, Wb1, bb1, Wb2, bb2, logits);
    k_softmax<<<Bc * Hc * Nc, 256, 0, stream>>>(logits);
    // 7-8. attn = aw@v  (+ fused rel-value einsum)
    k_attn1<<<BNc, 256, 0, stream>>>(logits, v_f, attn_f);
    k_attnrel<<<BNc, 128, 0, stream>>>(logits, pos_f, Wv1, bv1, Wv2, bv2, attn_f);
    // 9. updated summary + glob norm (+ output #2)
    k_us<<<BNc, 256, 0, stream>>>(attn_f, s_f, us_f, gn_f, dout2);
    // 10-12. gate path
    k_combined<<<BNTc, 256, 0, stream>>>(pb, gn_f, comb_bf);
    dim3 ggate(Dc / 64, BNTc / 64);
    k_gemm<2, 0><<<ggate, 256, 0, stream>>>(comb_bf, wg_bf, bg, gate_x, nullptr, BNTc, Dc, 2 * Dc);
    k_lnx<<<BNTc, 256, 0, stream>>>(gate_x, pb, us_f, gng, gnb, x_bf);
    // 13-14. FFN (WMMA)
    dim3 gff1(FFc / 64, BNTc / 64);
    k_gemm<1, 1><<<gff1, 256, 0, stream>>>(x_bf, wf1_bf, bf1, nullptr, h1_bf, BNTc, FFc, Dc);
    dim3 gff2(Dc / 64, BNTc / 64);
    k_gemm<0, 0><<<gff2, 256, 0, stream>>>(h1_bf, wf2_bf, bf2, dout1, nullptr, BNTc, Dc, FFc);
    // 15. final LayerNorm in place on d_out
    k_lnfinal<<<BNTc, 256, 0, stream>>>(dout1, gate_x, fng, fnb);
}
